// EdgeGroupImportanceModel_34256659153223
// MI455X (gfx1250) — matrix operand order repaired and checked by reading, not compiled, not run on hardware
//
#include <hip/hip_runtime.h>
#include <hip/hip_bf16.h>

typedef __attribute__((ext_vector_type(16))) _Float16 v16h;
typedef __attribute__((ext_vector_type(8)))  _Float16 v8h;
typedef __attribute__((ext_vector_type(8)))  float    v8f;

#define N_EDG   4096
#define E_DIMM  128
#define HIDD    256
#define NHEADS  4
#define TOPK    16384
#define N_GRP   131072
#define NNODES  2048

__device__ __forceinline__ float wave_sum32(float v){
#pragma unroll
  for (int o = 16; o > 0; o >>= 1) v += __shfl_xor(v, o, 32);
  return v;
}

// ----------------------------- utility kernels -----------------------------
__global__ void k_fill_f32(float* p, float v, int n){
  int i = blockIdx.x * blockDim.x + threadIdx.x;
  if (i < n) p[i] = v;
}
__global__ void k_to_half(const float* x, _Float16* y, int n){
  int i = blockIdx.x * blockDim.x + threadIdx.x;
  if (i < n) y[i] = (_Float16)x[i];
}
// W is [K][Nn] row-major f32 -> Wt is [Nn][K] row-major f16
__global__ void k_transpose_to_half(const float* W, _Float16* Wt, int K, int Nn){
  int id = blockIdx.x * blockDim.x + threadIdx.x;
  if (id >= K * Nn) return;
  int k = id / Nn, n = id % Nn;
  Wt[(size_t)n * K + k] = (_Float16)W[id];
}
__global__ void k_iota(int* p, int n){
  int i = blockIdx.x * blockDim.x + threadIdx.x;
  if (i < n) p[i] = i;
}
__global__ void k_extract_pairs(const int* eg, int* a, int* b, int n){
  int i = blockIdx.x * blockDim.x + threadIdx.x;
  if (i < n){ a[i] = eg[2 * i]; b[i] = eg[2 * i + 1]; }
}

// ----------------------------- WMMA GEMM (32x32 tile / wave) ---------------
// C[M,Nn] = act( concat(A0[idx0],A1[idx1]) @ Wt^T + bias + rowScale⊗colVec + residual )
// A0: [*, K0] f16 (rows selected by idx0, identity if null)
// A1: [*, K1] f16 (K1 may be 0)
// Wt: [Nn][K] f16 (pre-transposed weights), K = K0+K1, K0%32==0, K1%32==0,
// M%32==0, Nn%32==0. One wave32 computes a 32x32 tile: 2 A-frags x 2 B-frags
// -> 4 v_wmma_f32_16x16x32_f16 per K-step. The K loop is split at the concat
// boundary so the inner loops have no per-load address selection.
__device__ __forceinline__ v16h joinA(v8h lo, v8h hi){
  v16h a;
#pragma unroll
  for (int e = 0; e < 8; ++e){ a[e] = lo[e]; a[e + 8] = hi[e]; }
  return a;
}

__global__ void k_wmma_gemm(const _Float16* __restrict__ A0,
                            const _Float16* __restrict__ A1,
                            int K0, int K1,
                            const int* __restrict__ idx0,
                            const int* __restrict__ idx1,
                            const _Float16* __restrict__ Wt,
                            const float* __restrict__ bias,
                            const float* __restrict__ rowScale,
                            const float* __restrict__ colVec,
                            const float* __restrict__ residual,
                            float* __restrict__ Cf,
                            _Float16* __restrict__ Ch,
                            int M, int Nn, int act){
  const int K = K0 + K1;
  const int tilesN = Nn >> 5;
  const int tiles  = (M >> 5) * tilesN;
  const int wid  = (blockIdx.x * blockDim.x + threadIdx.x) >> 5;
  if (wid >= tiles) return;                       // uniform per wave
  const int lane = threadIdx.x & 31;
  const int tileM = wid / tilesN;
  const int tileN = wid % tilesN;
  const int r  = lane & 15;                       // M-row within sub-tile (A), N-col (B/C)
  const int hs = lane >> 4;                       // half-wave selector

  const int rowA0 = tileM * 32 + r;
  const int rowA1 = rowA0 + 16;
  // hoisted per-row base pointers (no selects in the hot loop)
  const _Float16* a0p = A0 + (size_t)(idx0 ? idx0[rowA0] : rowA0) * K0 + hs * 8;
  const _Float16* a1p = A0 + (size_t)(idx0 ? idx0[rowA1] : rowA1) * K0 + hs * 8;
  const _Float16* a0q = nullptr;
  const _Float16* a1q = nullptr;
  if (K1 > 0){
    a0q = A1 + (size_t)(idx1 ? idx1[rowA0] : rowA0) * K1 + hs * 8;
    a1q = A1 + (size_t)(idx1 ? idx1[rowA1] : rowA1) * K1 + hs * 8;
  }
  const _Float16* bbase0 = Wt + (size_t)(tileN * 32 + r) * K + hs * 8;
  const _Float16* bbase1 = bbase0 + (size_t)16 * K;

  v8f acc00 = {}, acc01 = {}, acc10 = {}, acc11 = {};
  for (int kk = 0; kk < K0; kk += 32){
    v16h a0 = joinA(*(const v8h*)(a0p + kk), *(const v8h*)(a0p + kk + 16));
    v16h a1 = joinA(*(const v8h*)(a1p + kk), *(const v8h*)(a1p + kk + 16));
    v16h b0 = joinA(*(const v8h*)(bbase0 + kk), *(const v8h*)(bbase0 + kk + 16));
    v16h b1 = joinA(*(const v8h*)(bbase1 + kk), *(const v8h*)(bbase1 + kk + 16));
    acc00 = __builtin_amdgcn_wmma_f32_16x16x32_f16(false, a0, false, b0, (short)0, acc00, false, false);
    acc01 = __builtin_amdgcn_wmma_f32_16x16x32_f16(false, a0, false, b1, (short)0, acc01, false, false);
    acc10 = __builtin_amdgcn_wmma_f32_16x16x32_f16(false, a1, false, b0, (short)0, acc10, false, false);
    acc11 = __builtin_amdgcn_wmma_f32_16x16x32_f16(false, a1, false, b1, (short)0, acc11, false, false);
  }
  for (int kk = 0; kk < K1; kk += 32){
    v16h a0 = joinA(*(const v8h*)(a0q + kk), *(const v8h*)(a0q + kk + 16));
    v16h a1 = joinA(*(const v8h*)(a1q + kk), *(const v8h*)(a1q + kk + 16));
    const _Float16* b0p = bbase0 + K0 + kk;
    const _Float16* b1p = bbase1 + K0 + kk;
    v16h b0 = joinA(*(const v8h*)(b0p), *(const v8h*)(b0p + 16));
    v16h b1 = joinA(*(const v8h*)(b1p), *(const v8h*)(b1p + 16));
    acc00 = __builtin_amdgcn_wmma_f32_16x16x32_f16(false, a0, false, b0, (short)0, acc00, false, false);
    acc01 = __builtin_amdgcn_wmma_f32_16x16x32_f16(false, a0, false, b1, (short)0, acc01, false, false);
    acc10 = __builtin_amdgcn_wmma_f32_16x16x32_f16(false, a1, false, b0, (short)0, acc10, false, false);
    acc11 = __builtin_amdgcn_wmma_f32_16x16x32_f16(false, a1, false, b1, (short)0, acc11, false, false);
  }

#pragma unroll
  for (int sm = 0; sm < 2; ++sm){
#pragma unroll
    for (int sn = 0; sn < 2; ++sn){
      const v8f acc = sm == 0 ? (sn == 0 ? acc00 : acc01)
                              : (sn == 0 ? acc10 : acc11);
#pragma unroll
      for (int j = 0; j < 8; ++j){
        const int m = tileM * 32 + sm * 16 + j + 8 * hs;
        const int n = tileN * 32 + sn * 16 + r;
        float v = acc[j];
        if (bias)     v += bias[n];
        if (rowScale) v += rowScale[m] * colVec[n];
        if (residual) v += residual[(size_t)m * Nn + n];
        if (act == 1)      v = fmaxf(v, 0.f);
        else if (act == 2) v = 0.5f * v * (1.f + erff(v * 0.70710678118654752f)); // exact gelu
        if (Cf) Cf[(size_t)m * Nn + n] = v;
        if (Ch) Ch[(size_t)m * Nn + n] = (_Float16)v;
      }
    }
  }
}

// ----------------------------- attention (flash, wave per row*head) --------
__global__ void k_attention(const float* __restrict__ qkv, _Float16* __restrict__ ctxH){
  const int wid  = (blockIdx.x * blockDim.x + threadIdx.x) >> 5;
  const int lane = threadIdx.x & 31;
  if (wid >= N_EDG * NHEADS) return;
  const int n = wid >> 2;
  const int h = wid & 3;
  const float scale = 0.17677669529663687f;   // 1/sqrt(32)
  const float qd = qkv[(size_t)n * 384 + h * 32 + lane];
  float mmax = -3.0e38f, lsum = 0.f, acc = 0.f;
  for (int m = 0; m < N_EDG; ++m){
    const float* rowp = qkv + (size_t)m * 384 + h * 32 + lane;
    float p = qd * rowp[128];                 // k
    p = wave_sum32(p);
    const float s = p * scale;
    const float mnew = fmaxf(mmax, s);
    const float corr = __expf(mmax - mnew);
    const float w    = __expf(s - mnew);
    lsum = lsum * corr + w;
    acc  = acc  * corr + w * rowp[256];       // v
    mmax = mnew;
  }
  ctxH[(size_t)n * 128 + h * 32 + lane] = (_Float16)(acc / lsum);
}

// ----------------------------- layernorm (wave per row) --------------------
__global__ void k_layernorm(const float* __restrict__ X,
                            const float* __restrict__ g,
                            const float* __restrict__ b,
                            _Float16* __restrict__ Y, int rows, int W){
  const int wid  = (blockIdx.x * blockDim.x + threadIdx.x) >> 5;
  const int lane = threadIdx.x & 31;
  if (wid >= rows) return;
  const float* x = X + (size_t)wid * W;
  float s = 0.f, ss = 0.f;
  for (int i = lane; i < W; i += 32){ float v = x[i]; s += v; ss += v * v; }
  s = wave_sum32(s); ss = wave_sum32(ss);
  const float mean = s / W;
  const float var  = ss / W - mean * mean;
  const float inv  = rsqrtf(var + 1e-5f);
  _Float16* y = Y + (size_t)wid * W;
  for (int i = lane; i < W; i += 32)
    y[i] = (_Float16)((x[i] - mean) * inv * g[i] + b[i]);
}

// ----------------------------- degree histogram ----------------------------
__global__ void k_degree(const int* __restrict__ ei, float* __restrict__ deg, int nE){
  int e = blockIdx.x * blockDim.x + threadIdx.x;
  if (e >= nE) return;
  atomicAdd(&deg[ei[e]], 1.0f);
  atomicAdd(&deg[ei[nE + e]], 1.0f);
}

// ------------- fused importance: compat2 + struct + comb + sigmoid ---------
__global__ void k_importance(const _Float16* __restrict__ h1,
                             const float* __restrict__ cw2, const float* __restrict__ cb2,
                             const float* __restrict__ deg, const int* __restrict__ ei,
                             const int* __restrict__ gi,  const int* __restrict__ gj,
                             const float* __restrict__ sw1, const float* __restrict__ sb1,
                             const float* __restrict__ sw2, const float* __restrict__ sb2,
                             const float* __restrict__ mw1, const float* __restrict__ mb1,
                             const float* __restrict__ mw2, const float* __restrict__ mb2,
                             float* __restrict__ imp, int G){
  int g = blockIdx.x * blockDim.x + threadIdx.x;
  if (g >= G) return;
  const _Float16* hrow = h1 + (size_t)g * 256;
  float c0 = cb2[0], c1 = cb2[1], c2 = cb2[2], c3 = cb2[3];
  for (int i = 0; i < 256; ++i){
    const float hv = (float)hrow[i];
    const float* w = cw2 + i * 4;
    c0 += hv * w[0]; c1 += hv * w[1]; c2 += hv * w[2]; c3 += hv * w[3];
  }
  c0 = fmaxf(c0, 0.f); c1 = fmaxf(c1, 0.f); c2 = fmaxf(c2, 0.f); c3 = fmaxf(c3, 0.f);
  const int ga = gi[g], gb = gj[g];
  const float f0 = deg[ei[ga]],           f1 = deg[ei[N_EDG + ga]];
  const float f2 = deg[ei[gb]],           f3 = deg[ei[N_EDG + gb]];
  float sacc = sb2[0];
  for (int j = 0; j < 64; ++j){
    float t = sb1[j] + f0 * sw1[j] + f1 * sw1[64 + j] + f2 * sw1[128 + j] + f3 * sw1[192 + j];
    sacc += fmaxf(t, 0.f) * sw2[j];
  }
  float logit = mb2[0];
  for (int j = 0; j < 128; ++j){
    float t = mb1[j] + c0 * mw1[j] + c1 * mw1[128 + j] + c2 * mw1[256 + j]
                     + c3 * mw1[384 + j] + sacc * mw1[512 + j];
    logit += fmaxf(t, 0.f) * mw2[j];
  }
  imp[g] = 1.f / (1.f + __expf(-logit));
}

// ----------------------------- bitonic sort stage --------------------------
// ordering: key descending, index ascending on ties (matches lax.top_k)
__global__ void k_bitonic(float* __restrict__ key, int* __restrict__ idx, int j, int k, int n){
  int i = blockIdx.x * blockDim.x + threadIdx.x;
  if (i >= n) return;
  int p = i ^ j;
  if (p <= i) return;
  float ka = key[i], kb = key[p];
  int   ia = idx[i], ib = idx[p];
  bool aLessB = (ka > kb) || (ka == kb && ia < ib);
  bool bLessA = (kb > ka) || (kb == ka && ib < ia);
  bool up = ((i & k) == 0);
  bool doswap = up ? bLessA : aLessB;
  if (doswap){ key[i] = kb; key[p] = ka; idx[i] = ib; idx[p] = ia; }
}

__global__ void k_gather_top(const float* __restrict__ keys, const int* __restrict__ idx,
                             const int* __restrict__ eg,
                             float* __restrict__ out_tv, float* __restrict__ out_tg,
                             int* __restrict__ ti, int* __restrict__ tj){
  int t = blockIdx.x * blockDim.x + threadIdx.x;
  if (t >= TOPK) return;
  out_tv[t] = keys[t];
  int id = idx[t];
  int a = eg[2 * id], b = eg[2 * id + 1];
  ti[t] = a; tj[t] = b;
  out_tg[2 * t] = (float)a; out_tg[2 * t + 1] = (float)b;
}

// ----------------------------- scatter / normalize -------------------------
__global__ void k_count(const int* __restrict__ ti, const int* __restrict__ tj,
                        float* __restrict__ cnt){
  int t = blockIdx.x * blockDim.x + threadIdx.x;
  if (t >= TOPK) return;
  atomicAdd(&cnt[ti[t]], 1.0f);
  atomicAdd(&cnt[tj[t]], 1.0f);
}
__global__ void k_scatter_add(const float* __restrict__ ge, const int* __restrict__ ti,
                              const int* __restrict__ tj, float* __restrict__ ctx2){
  int id = blockIdx.x * blockDim.x + threadIdx.x;
  if (id >= TOPK * HIDD) return;
  int t = id >> 8, c = id & 255;
  float v = ge[id];
  atomicAdd(&ctx2[(size_t)ti[t] * HIDD + c], v);
  atomicAdd(&ctx2[(size_t)tj[t] * HIDD + c], v);
}
__global__ void k_norm_ctx2(const float* __restrict__ ctx2, const float* __restrict__ cnt,
                            _Float16* __restrict__ out){
  int id = blockIdx.x * blockDim.x + threadIdx.x;
  if (id >= N_EDG * HIDD) return;
  int nrow = id >> 8;
  out[id] = (_Float16)(ctx2[id] / fmaxf(cnt[nrow], 1.0f));
}

// ============================ host side =====================================
extern "C" void kernel_launch(void* const* d_in, const int* in_sizes, int n_in,
                              void* d_out, int out_size, void* d_ws, size_t ws_size,
                              hipStream_t stream){
  (void)in_sizes; (void)n_in; (void)out_size; (void)ws_size;
  const float* emb_in   = (const float*)d_in[0];
  const int*   edge_idx = (const int*)d_in[1];      // (2, 4096)
  const int*   edge_grp = (const int*)d_in[2];      // (131072, 2)
  const float* in_proj_w = (const float*)d_in[3];
  const float* in_proj_b = (const float*)d_in[4];
  const float* out_w     = (const float*)d_in[5];
  const float* out_b     = (const float*)d_in[6];
  const float* compat_w1 = (const float*)d_in[7];
  const float* compat_b1 = (const float*)d_in[8];
  const float* compat_w2 = (const float*)d_in[9];
  const float* compat_b2 = (const float*)d_in[10];
  const float* str_w1    = (const float*)d_in[11];
  const float* str_b1    = (const float*)d_in[12];
  const float* str_w2    = (const float*)d_in[13];
  const float* str_b2    = (const float*)d_in[14];
  const float* comb_w1   = (const float*)d_in[15];
  const float* comb_b1   = (const float*)d_in[16];
  const float* comb_w2   = (const float*)d_in[17];
  const float* comb_b2   = (const float*)d_in[18];
  const float* g_w1      = (const float*)d_in[19];
  const float* g_b1      = (const float*)d_in[20];
  const float* g_ln1_g   = (const float*)d_in[21];
  const float* g_ln1_b   = (const float*)d_in[22];
  const float* g_w2      = (const float*)d_in[23];
  const float* g_b2      = (const float*)d_in[24];
  const float* g_ln2_g   = (const float*)d_in[25];
  const float* g_ln2_b   = (const float*)d_in[26];
  const float* mod_w     = (const float*)d_in[27];  // (257, 256)
  const float* mod_b     = (const float*)d_in[28];
  const float* ref_w1    = (const float*)d_in[29];
  const float* ref_b1    = (const float*)d_in[30];
  const float* ref_ln_g  = (const float*)d_in[31];
  const float* ref_ln_b  = (const float*)d_in[32];
  const float* ref_w2    = (const float*)d_in[33];
  const float* ref_b2    = (const float*)d_in[34];

  // output layout: refined(4096*128) | group_emb(16384*256) | top_vals(16384) | tg(16384*2)
  float* out_refined = (float*)d_out;
  float* out_ge      = out_refined + (size_t)N_EDG * E_DIMM;
  float* out_tv      = out_ge + (size_t)TOPK * HIDD;
  float* out_tg      = out_tv + TOPK;

  // workspace carve-up
  size_t off = 0;
  auto alloc = [&](size_t bytes) -> char* {
    off = (off + 255) & ~(size_t)255;
    char* p = (char*)d_ws + off;
    off += bytes;
    return p;
  };
  _Float16* embH    = (_Float16*)alloc((size_t)N_EDG * E_DIMM * 2);
  _Float16* inprojT = (_Float16*)alloc((size_t)384 * 128 * 2);
  _Float16* outT    = (_Float16*)alloc((size_t)128 * 128 * 2);
  _Float16* compatT = (_Float16*)alloc((size_t)256 * 256 * 2);
  _Float16* g1T     = (_Float16*)alloc((size_t)512 * 256 * 2);
  _Float16* g2T     = (_Float16*)alloc((size_t)256 * 512 * 2);
  _Float16* modT    = (_Float16*)alloc((size_t)256 * 256 * 2);
  _Float16* ref1T   = (_Float16*)alloc((size_t)256 * 384 * 2);
  _Float16* ref2T   = (_Float16*)alloc((size_t)128 * 256 * 2);
  float*    qkv     = (float*)alloc((size_t)N_EDG * 384 * 4);
  _Float16* ctxH    = (_Float16*)alloc((size_t)N_EDG * 128 * 2);
  float*    embOut  = (float*)alloc((size_t)N_EDG * 128 * 4);
  _Float16* embH2   = (_Float16*)alloc((size_t)N_EDG * 128 * 2);
  int*      giArr   = (int*)alloc((size_t)N_GRP * 4);
  int*      gjArr   = (int*)alloc((size_t)N_GRP * 4);
  _Float16* h1      = (_Float16*)alloc((size_t)N_GRP * 256 * 2);
  float*    deg     = (float*)alloc((size_t)NNODES * 4);
  float*    imp     = (float*)alloc((size_t)N_GRP * 4);   // sort keys (in place)
  int*      sidx    = (int*)alloc((size_t)N_GRP * 4);
  int*      ti      = (int*)alloc((size_t)TOPK * 4);
  int*      tj      = (int*)alloc((size_t)TOPK * 4);
  float*    gh      = (float*)alloc((size_t)TOPK * 512 * 4);
  _Float16* ghH     = (_Float16*)alloc((size_t)TOPK * 512 * 2);
  float*    h2f     = (float*)alloc((size_t)TOPK * 256 * 4);
  _Float16* h2H     = (_Float16*)alloc((size_t)TOPK * 256 * 2);
  float*    ctx2    = (float*)alloc((size_t)N_EDG * HIDD * 4);
  float*    cnt     = (float*)alloc((size_t)N_EDG * 4);
  _Float16* ctx2H   = (_Float16*)alloc((size_t)N_EDG * HIDD * 2);
  float*    refh    = (float*)alloc((size_t)N_EDG * 256 * 4);
  _Float16* refhH   = (_Float16*)alloc((size_t)N_EDG * 256 * 2);

  auto grid1 = [](int n){ return (n + 255) / 256; };
  auto launch_gemm = [&](const _Float16* A0, const _Float16* A1, int K0, int K1,
                         const int* i0, const int* i1, const _Float16* Wt,
                         const float* bias, const float* rs, const float* cv,
                         const float* res, float* Cf, _Float16* Ch,
                         int M, int Nn, int act){
    int waves = (M / 32) * (Nn / 32);
    k_wmma_gemm<<<grid1(waves * 32), 256, 0, stream>>>(
        A0, A1, K0, K1, i0, i1, Wt, bias, rs, cv, res, Cf, Ch, M, Nn, act);
  };

  // ---- prepare fp16 activations & transposed fp16 weights ----
  k_to_half<<<grid1(N_EDG * E_DIMM), 256, 0, stream>>>(emb_in, embH, N_EDG * E_DIMM);
  k_transpose_to_half<<<grid1(128 * 384), 256, 0, stream>>>(in_proj_w, inprojT, 128, 384);
  k_transpose_to_half<<<grid1(128 * 128), 256, 0, stream>>>(out_w,     outT,    128, 128);
  k_transpose_to_half<<<grid1(256 * 256), 256, 0, stream>>>(compat_w1, compatT, 256, 256);
  k_transpose_to_half<<<grid1(256 * 512), 256, 0, stream>>>(g_w1,      g1T,     256, 512);
  k_transpose_to_half<<<grid1(512 * 256), 256, 0, stream>>>(g_w2,      g2T,     512, 256);
  k_transpose_to_half<<<grid1(256 * 256), 256, 0, stream>>>(mod_w,     modT,    256, 256); // rows 0..255
  k_transpose_to_half<<<grid1(384 * 256), 256, 0, stream>>>(ref_w1,    ref1T,   384, 256);
  k_transpose_to_half<<<grid1(256 * 128), 256, 0, stream>>>(ref_w2,    ref2T,   256, 128);

  // ---- attention block ----
  launch_gemm(embH, nullptr, 128, 0, nullptr, nullptr, inprojT, in_proj_b,
              nullptr, nullptr, nullptr, qkv, nullptr, N_EDG, 384, 0);
  k_attention<<<grid1(N_EDG * NHEADS * 32), 256, 0, stream>>>(qkv, ctxH);
  launch_gemm(ctxH, nullptr, 128, 0, nullptr, nullptr, outT, out_b,
              nullptr, nullptr, nullptr, embOut, embH2, N_EDG, 128, 0);

  // ---- compat layer1 (gather-fused pair GEMM), fp16 out ----
  k_extract_pairs<<<grid1(N_GRP), 256, 0, stream>>>(edge_grp, giArr, gjArr, N_GRP);
  launch_gemm(embH2, embH2, 128, 128, giArr, gjArr, compatT, compat_b1,
              nullptr, nullptr, nullptr, nullptr, h1, N_GRP, 256, 1);

  // ---- degree + fused importance ----
  k_fill_f32<<<grid1(NNODES), 256, 0, stream>>>(deg, 0.f, NNODES);
  k_degree<<<grid1(N_EDG), 256, 0, stream>>>(edge_idx, deg, N_EDG);
  k_importance<<<grid1(N_GRP), 256, 0, stream>>>(h1, compat_w2, compat_b2,
      deg, edge_idx, giArr, gjArr, str_w1, str_b1, str_w2, str_b2,
      comb_w1, comb_b1, comb_w2, comb_b2, imp, N_GRP);

  // ---- exact sorted top-k via bitonic sort (desc key, asc idx) ----
  k_iota<<<grid1(N_GRP), 256, 0, stream>>>(sidx, N_GRP);
  for (int k = 2; k <= N_GRP; k <<= 1)
    for (int j = k >> 1; j > 0; j >>= 1)
      k_bitonic<<<grid1(N_GRP), 256, 0, stream>>>(imp, sidx, j, k, N_GRP);
  k_gather_top<<<grid1(TOPK), 256, 0, stream>>>(imp, sidx, edge_grp,
                                                out_tv, out_tg, ti, tj);

  // ---- group MLP ----
  launch_gemm(embH2, embH2, 128, 128, ti, tj, g1T, g_b1,
              nullptr, nullptr, nullptr, gh, nullptr, TOPK, 512, 2);   // gelu
  k_layernorm<<<grid1(TOPK * 32), 256, 0, stream>>>(gh, g_ln1_g, g_ln1_b, ghH, TOPK, 512);
  launch_gemm(ghH, nullptr, 512, 0, nullptr, nullptr, g2T, g_b2,
              nullptr, nullptr, nullptr, h2f, nullptr, TOPK, 256, 1);  // relu
  k_layernorm<<<grid1(TOPK * 32), 256, 0, stream>>>(h2f, g_ln2_g, g_ln2_b, h2H, TOPK, 256);
  // mod layer: K=257 folded as rank-1 epilogue top_vals[m] * mod_w[256][n]
  launch_gemm(h2H, nullptr, 256, 0, nullptr, nullptr, modT, mod_b,
              out_tv, mod_w + (size_t)256 * 256, nullptr,
              out_ge, nullptr, TOPK, 256, 1);                          // relu -> group_emb

  // ---- scatter mean ----
  k_fill_f32<<<grid1(N_EDG * HIDD), 256, 0, stream>>>(ctx2, 0.f, N_EDG * HIDD);
  k_fill_f32<<<grid1(N_EDG), 256, 0, stream>>>(cnt, 0.f, N_EDG);
  k_count<<<grid1(TOPK), 256, 0, stream>>>(ti, tj, cnt);
  k_scatter_add<<<grid1(TOPK * HIDD), 256, 0, stream>>>(out_ge, ti, tj, ctx2);
  k_norm_ctx2<<<grid1(N_EDG * HIDD), 256, 0, stream>>>(ctx2, cnt, ctx2H);

  // ---- refinement ----
  launch_gemm(embH2, ctx2H, 128, 256, nullptr, nullptr, ref1T, ref_b1,
              nullptr, nullptr, nullptr, refh, nullptr, N_EDG, 256, 1);
  k_layernorm<<<grid1(N_EDG * 32), 256, 0, stream>>>(refh, ref_ln_g, ref_ln_b, refhH, N_EDG, 256);
  launch_gemm(refhH, nullptr, 256, 0, nullptr, nullptr, ref2T, ref_b2,
              nullptr, nullptr, embOut, out_refined, nullptr, N_EDG, 128, 0); // +residual
}
